// CrossAttention_51702816309660
// MI455X (gfx1250) — matrix operand order repaired and checked by reading, not compiled, not run on hardware
//
#include <hip/hip_runtime.h>

// ---------------------------------------------------------------------------
// CDNA5 (gfx1250) cross-attention: bf16 WMMA everywhere, f32 accumulate,
// flash-style streaming softmax (attn matrix never materialized).
// Round 3: 4-deep B-fragment preload (distinct live ranges -> partial
// s_wait_dscnt, load/WMMA overlap), vectorized ushort4/float4 tile fills.
// ---------------------------------------------------------------------------

typedef __attribute__((ext_vector_type(16))) __bf16 v16bf;
typedef __attribute__((ext_vector_type(8)))  float  v8f;

#define N_HEADS 16
#define DHEAD   64
#define NORM_EPS 1e-12f
#define MAX_SCALE_MUL 4.605170185988091f   // log(100)

__device__ __forceinline__ float bf2f(__bf16 x) {
  unsigned short u = __builtin_bit_cast(unsigned short, x);
  return __builtin_bit_cast(float, (unsigned)u << 16);
}
__device__ __forceinline__ __bf16 f2bf(float f) {
  unsigned u = __builtin_bit_cast(unsigned, f);
  unsigned r;
  if ((u & 0x7f800000u) == 0x7f800000u) r = u;        // Inf/NaN: truncate
  else r = u + 0x7fffu + ((u >> 16) & 1u);            // round-to-nearest-even
  return __builtin_bit_cast(__bf16, (unsigned short)(r >> 16));
}
__device__ __forceinline__ unsigned short f2bfu(float f) {
  return __builtin_bit_cast(unsigned short, f2bf(f));
}

// --- WMMA fragment loaders (layouts per CDNA5 ISA §7.12.2) ------
// A matrix 16x32 bf16: lanes 0-15 -> M=0..15 (K 0..7,16..23), lanes 16-31
// same M, K 8..15,24..31.  K runs are contiguous -> wide DS loads.
__device__ __forceinline__ v16bf load_a_frag(const __bf16* base, int ldm,
                                             int row0, int col0, int lane) {
  const int half = lane >> 4, m = lane & 15;
  const __bf16* p = base + (size_t)(row0 + m) * ldm + col0 + 8 * half;
  v16bf a;
#pragma unroll
  for (int i = 0; i < 8; ++i) a[i]     = p[i];        // K = 8*half + {0..7}
#pragma unroll
  for (int i = 0; i < 8; ++i) a[8 + i] = p[16 + i];   // K = 16+8*half + {0..7}
  return a;
}
// B matrix 32x16 (KxN): lane = N column; same K order as A fragment:
// lanes 0-15 K 0..7,16..23, lanes 16-31 K 8..15,24..31. LDS tile is N-major
// (LDS[n][k]).
__device__ __forceinline__ v16bf load_b_frag_nmajor(const __bf16* base, int ldm,
                                                    int nrow0, int col0, int lane) {
  const int half = lane >> 4, n = lane & 15;
  const __bf16* p = base + (size_t)(nrow0 + n) * ldm + col0 + 8 * half;
  v16bf b;
#pragma unroll
  for (int v = 0; v < 8; ++v) b[v] = p[v];
#pragma unroll
  for (int v = 0; v < 8; ++v) b[8 + v] = p[16 + v];
  return b;
}

// ---------------------------------------------------------------------------
// GEMM:  C[m][n] = sum_k A[m][k] * W[n][k]   (torch Linear, y = x @ W^T)
// 64x64 output tile per workgroup, 4 waves, each wave owns a 16-row strip.
// ABF16: A operand already bf16 (workspace); OUTF32: f32 output (+bias).
// ---------------------------------------------------------------------------
template <bool ABF16, bool OUTF32>
__global__ __launch_bounds__(128)
void gemm_wt_kernel(const void* __restrict__ Av, const float* __restrict__ W,
                    const float* __restrict__ bias, void* __restrict__ Outv,
                    int M, int N, int K) {
  const int m0 = blockIdx.x * 64, n0 = blockIdx.y * 64;
  const int tid = threadIdx.x, wv = tid >> 5, lane = tid & 31;
  const int half = lane >> 4, ln = lane & 15;

  __shared__ __bf16 As[64][36];   // 72B rows: 8B aligned, 18-dword stride
  __shared__ __bf16 Ws[64][36];

  v8f acc[4] = {{}, {}, {}, {}};

  for (int kc = 0; kc < K; kc += 32) {
    __syncthreads();
#pragma unroll
    for (int idx = tid; idx < 64 * 8; idx += 128) {    // 4-element quads
      const int r = idx >> 3, c4 = (idx & 7) * 4;
      if (ABF16) {
        *(ushort4*)&As[r][c4] =
            *(const ushort4*)((const __bf16*)Av + (size_t)(m0 + r) * K + kc + c4);
      } else {
        const float4 a4 =
            *(const float4*)((const float*)Av + (size_t)(m0 + r) * K + kc + c4);
        ushort4 u; u.x = f2bfu(a4.x); u.y = f2bfu(a4.y);
                   u.z = f2bfu(a4.z); u.w = f2bfu(a4.w);
        *(ushort4*)&As[r][c4] = u;
      }
      const float4 w4 = *(const float4*)(W + (size_t)(n0 + r) * K + kc + c4);
      ushort4 uw; uw.x = f2bfu(w4.x); uw.y = f2bfu(w4.y);
                  uw.z = f2bfu(w4.z); uw.w = f2bfu(w4.w);
      *(ushort4*)&Ws[r][c4] = uw;
    }
    __syncthreads();

    const v16bf a  = load_a_frag(&As[0][0], 36, wv * 16, 0, lane);
    const v16bf b0 = load_b_frag_nmajor(&Ws[0][0], 36, 0,  0, lane);
    const v16bf b1 = load_b_frag_nmajor(&Ws[0][0], 36, 16, 0, lane);
    const v16bf b2 = load_b_frag_nmajor(&Ws[0][0], 36, 32, 0, lane);
    const v16bf b3 = load_b_frag_nmajor(&Ws[0][0], 36, 48, 0, lane);
    acc[0] = __builtin_amdgcn_wmma_f32_16x16x32_bf16(false, a, false, b0, (short)0, acc[0], false, false);
    acc[1] = __builtin_amdgcn_wmma_f32_16x16x32_bf16(false, a, false, b1, (short)0, acc[1], false, false);
    acc[2] = __builtin_amdgcn_wmma_f32_16x16x32_bf16(false, a, false, b2, (short)0, acc[2], false, false);
    acc[3] = __builtin_amdgcn_wmma_f32_16x16x32_bf16(false, a, false, b3, (short)0, acc[3], false, false);
  }

#pragma unroll
  for (int nt = 0; nt < 4; ++nt) {
#pragma unroll
    for (int r = 0; r < 8; ++r) {
      const size_t row = (size_t)m0 + wv * 16 + r + 8 * half;
      const size_t col = (size_t)n0 + nt * 16 + ln;
      const float v = acc[nt][r];
      if (OUTF32) ((float*)Outv)[row * N + col] = v + (bias ? bias[col] : 0.f);
      else        ((__bf16*)Outv)[row * N + col] = f2bf(v);
    }
  }
}

// ---------------------------------------------------------------------------
// Q scaling: q <- q / max(||q||_2 over d, eps) * exp(min(scale_mul[h],log100))
// One wave per (row, head) group of 64 elements.
// ---------------------------------------------------------------------------
__global__ __launch_bounds__(256)
void qscale_kernel(__bf16* __restrict__ Q, const float* __restrict__ scale_mul,
                   int rows /* B*L */, int C) {
  const int wave = (blockIdx.x * blockDim.x + threadIdx.x) >> 5;
  const int lane = threadIdx.x & 31;
  if (wave >= rows * N_HEADS) return;
  const int h = wave % N_HEADS, row = wave / N_HEADS;
  __bf16* p = Q + (size_t)row * C + h * DHEAD;
  const float x0 = bf2f(p[lane]), x1 = bf2f(p[lane + 32]);
  float ss = x0 * x0 + x1 * x1;
#pragma unroll
  for (int mk = 16; mk >= 1; mk >>= 1) ss += __shfl_xor(ss, mk, 32);
  const float nrm = fmaxf(sqrtf(ss), NORM_EPS);
  const float sm = __expf(fminf(scale_mul[h], MAX_SCALE_MUL));
  const float s = sm / nrm;
  p[lane]      = f2bf(x0 * s);
  p[lane + 32] = f2bf(x1 * s);
}

// ---------------------------------------------------------------------------
// K scaling (faithful to source): k permuted to (B,H,d,Lc) BEFORE normalize,
// so the norm runs over the Lc axis: per (b, channel c) column of K.
// ---------------------------------------------------------------------------
__global__ __launch_bounds__(256)
void kscale_kernel(__bf16* __restrict__ K, int Lc, int C) {
  const int b = blockIdx.y;
  const int c0 = blockIdx.x * 64;
  const int tc = threadIdx.x & 63, tg = threadIdx.x >> 6;   // 4 groups
  __bf16* base = K + (size_t)b * Lc * C + c0;
  float ss = 0.f;
  for (int m = tg; m < Lc; m += 4) {
    const float v = bf2f(base[(size_t)m * C + tc]);
    ss += v * v;
  }
  __shared__ float red[4][64];
  red[tg][tc] = ss;
  __syncthreads();
  const float tot = red[0][tc] + red[1][tc] + red[2][tc] + red[3][tc];
  const float s = 1.0f / fmaxf(sqrtf(tot), NORM_EPS);
  for (int m = tg; m < Lc; m += 4) {
    base[(size_t)m * C + tc] = f2bf(bf2f(base[(size_t)m * C + tc]) * s);
  }
}

// ---------------------------------------------------------------------------
// Flash attention: workgroup = 64 query rows of one (b,h); 4 waves, each owns
// a 16-row strip. Stream Lc in 64-wide tiles. S and O live in WMMA C/D
// fragment layout: element (M = r + 8*(lane/16), N = lane%16); row-wise
// softmax reductions are 16-lane shfl_xor within each lane-half.
// Q fragments held in registers for the whole Lc loop; V tile transposed in
// LDS so both WMMA operand patterns are contiguous reads.
// ---------------------------------------------------------------------------
__global__ __launch_bounds__(128)
void attn_kernel(const __bf16* __restrict__ Q, const __bf16* __restrict__ K,
                 const __bf16* __restrict__ V, const float* __restrict__ bias,
                 __bf16* __restrict__ AO, int B, int L, int Lc, int C) {
  const int b = blockIdx.z, h = blockIdx.y, lt = blockIdx.x;
  const int tid = threadIdx.x, wv = tid >> 5, lane = tid & 31;
  const int half = lane >> 4, ln = lane & 15;

  __shared__ __bf16 Qs[64][68];
  __shared__ __bf16 Ks[64][68];       // Ks[m][dd]
  __shared__ __bf16 Vt[64][68];       // Vt[dd][m]  (transposed V tile)
  __shared__ __bf16 Ps[4][16][68];    // per-wave probability strip (relayout)

  const size_t qbase = ((size_t)b * L + lt * 64) * C + h * DHEAD;
#pragma unroll
  for (int idx = tid; idx < 64 * 16; idx += 128) {
    const int r = idx >> 4, c4 = (idx & 15) * 4;
    *(ushort4*)&Qs[r][c4] = *(const ushort4*)(Q + qbase + (size_t)r * C + c4);
  }
  __syncthreads();

  // Hoist Q A-fragments: invariant across the whole Lc loop.
  const v16bf qa0 = load_a_frag(&Qs[0][0], 68, wv * 16, 0,  lane);
  const v16bf qa1 = load_a_frag(&Qs[0][0], 68, wv * 16, 32, lane);

  v8f Ofrag[4] = {{}, {}, {}, {}};
  float rowmax[8], rowsum[8];
#pragma unroll
  for (int r = 0; r < 8; ++r) { rowmax[r] = -INFINITY; rowsum[r] = 0.f; }

  const size_t kvbase = (size_t)b * Lc * C + h * DHEAD;
  const float* brow0 =
      bias + ((size_t)h * L + (size_t)lt * 64 + wv * 16) * Lc;

  for (int m0 = 0; m0 < Lc; m0 += 64) {
#pragma unroll
    for (int idx = tid; idx < 64 * 16; idx += 128) {
      const int r = idx >> 4, c4 = (idx & 15) * 4;   // r = m row, c4 = dd quad
      const size_t g = kvbase + (size_t)(m0 + r) * C + c4;
      *(ushort4*)&Ks[r][c4] = *(const ushort4*)(K + g);
      const ushort4 vv = *(const ushort4*)(V + g);   // transpose on LDS store
      Vt[c4 + 0][r] = __builtin_bit_cast(__bf16, (unsigned short)vv.x);
      Vt[c4 + 1][r] = __builtin_bit_cast(__bf16, (unsigned short)vv.y);
      Vt[c4 + 2][r] = __builtin_bit_cast(__bf16, (unsigned short)vv.z);
      Vt[c4 + 3][r] = __builtin_bit_cast(__bf16, (unsigned short)vv.w);
    }
    // prefetch next K/V tile rows (emits global_prefetch_b8)
    if (m0 + 64 < Lc && tid < 64) {
      __builtin_prefetch(&K[kvbase + (size_t)(m0 + 64 + tid) * C], 0, 1);
      __builtin_prefetch(&V[kvbase + (size_t)(m0 + 64 + tid) * C], 0, 1);
    }
    __syncthreads();

    // ---- S = Q @ Kn^T  (16 x 64 per wave), f32 accumulate ----
    v8f st[4] = {{}, {}, {}, {}};
#pragma unroll
    for (int kc = 0; kc < DHEAD; kc += 32) {
      const v16bf a = (kc == 0) ? qa0 : qa1;
      const v16bf b0 = load_b_frag_nmajor(&Ks[0][0], 68, 0,  kc, lane);
      const v16bf b1 = load_b_frag_nmajor(&Ks[0][0], 68, 16, kc, lane);
      const v16bf b2 = load_b_frag_nmajor(&Ks[0][0], 68, 32, kc, lane);
      const v16bf b3 = load_b_frag_nmajor(&Ks[0][0], 68, 48, kc, lane);
      st[0] = __builtin_amdgcn_wmma_f32_16x16x32_bf16(false, a, false, b0, (short)0, st[0], false, false);
      st[1] = __builtin_amdgcn_wmma_f32_16x16x32_bf16(false, a, false, b1, (short)0, st[1], false, false);
      st[2] = __builtin_amdgcn_wmma_f32_16x16x32_bf16(false, a, false, b2, (short)0, st[2], false, false);
      st[3] = __builtin_amdgcn_wmma_f32_16x16x32_bf16(false, a, false, b3, (short)0, st[3], false, false);
    }
    // ---- + bias ----
    const float* brow = brow0 + m0;
#pragma unroll
    for (int nt = 0; nt < 4; ++nt)
#pragma unroll
      for (int r = 0; r < 8; ++r)
        st[nt][r] += brow[(size_t)(r + 8 * half) * Lc + nt * 16 + ln];

    // ---- online softmax update ----
#pragma unroll
    for (int r = 0; r < 8; ++r) {
      float mt = fmaxf(fmaxf(st[0][r], st[1][r]), fmaxf(st[2][r], st[3][r]));
#pragma unroll
      for (int mk = 8; mk >= 1; mk >>= 1) mt = fmaxf(mt, __shfl_xor(mt, mk, 16));
      const float nm = fmaxf(rowmax[r], mt);
      const float corr = __expf(rowmax[r] - nm);
      rowmax[r] = nm;
      float ps = 0.f;
#pragma unroll
      for (int nt = 0; nt < 4; ++nt) {
        const float p = __expf(st[nt][r] - nm);
        ps += p;
        Ps[wv][r + 8 * half][nt * 16 + ln] = f2bf(p);   // relayout via LDS
      }
#pragma unroll
      for (int mk = 8; mk >= 1; mk >>= 1) ps += __shfl_xor(ps, mk, 16);
      rowsum[r] = rowsum[r] * corr + ps;
#pragma unroll
      for (int dt = 0; dt < 4; ++dt) Ofrag[dt][r] *= corr;
    }
    __syncthreads();   // Ps visible; Ks fully consumed

    // ---- O += P @ V ----  (P A-fragment hoisted across dt)
#pragma unroll
    for (int kc = 0; kc < 64; kc += 32) {
      const v16bf a  = load_a_frag(&Ps[wv][0][0], 68, 0, kc, lane);
      const v16bf b0 = load_b_frag_nmajor(&Vt[0][0], 68, 0,  kc, lane);
      const v16bf b1 = load_b_frag_nmajor(&Vt[0][0], 68, 16, kc, lane);
      const v16bf b2 = load_b_frag_nmajor(&Vt[0][0], 68, 32, kc, lane);
      const v16bf b3 = load_b_frag_nmajor(&Vt[0][0], 68, 48, kc, lane);
      Ofrag[0] = __builtin_amdgcn_wmma_f32_16x16x32_bf16(false, a, false, b0, (short)0, Ofrag[0], false, false);
      Ofrag[1] = __builtin_amdgcn_wmma_f32_16x16x32_bf16(false, a, false, b1, (short)0, Ofrag[1], false, false);
      Ofrag[2] = __builtin_amdgcn_wmma_f32_16x16x32_bf16(false, a, false, b2, (short)0, Ofrag[2], false, false);
      Ofrag[3] = __builtin_amdgcn_wmma_f32_16x16x32_bf16(false, a, false, b3, (short)0, Ofrag[3], false, false);
    }
    __syncthreads();   // before next tile overwrites Ks/Vt
  }

  // ---- epilogue: O / rowsum -> bf16 attention-output buffer ----
  float inv[8];
#pragma unroll
  for (int r = 0; r < 8; ++r) inv[r] = 1.0f / rowsum[r];
#pragma unroll
  for (int dt = 0; dt < 4; ++dt)
#pragma unroll
    for (int r = 0; r < 8; ++r) {
      const size_t row = (size_t)b * L + lt * 64 + wv * 16 + r + 8 * half;
      AO[row * C + h * DHEAD + dt * 16 + ln] = f2bf(Ofrag[dt][r] * inv[r]);
    }
}

// ---------------------------------------------------------------------------
extern "C" void kernel_launch(void* const* d_in, const int* in_sizes, int n_in,
                              void* d_out, int out_size, void* d_ws, size_t ws_size,
                              hipStream_t stream) {
  const int B = 4, L = 1024, Lc = 2048, C = 1024;
  const float* x         = (const float*)d_in[0];
  const float* context   = (const float*)d_in[1];
  const float* attn_bias = (const float*)d_in[2];
  const float* Wq        = (const float*)d_in[3];
  const float* Wk        = (const float*)d_in[4];
  const float* Wv        = (const float*)d_in[5];
  const float* Wp        = (const float*)d_in[6];
  const float* bp        = (const float*)d_in[7];
  const float* scale_mul = (const float*)d_in[8];
  float* out = (float*)d_out;

  // workspace layout (bf16): Q | K | V | attn-out  (48 MB total)
  char* ws = (char*)d_ws;
  __bf16* Qb  = (__bf16*)ws;
  __bf16* Kb  = Qb + (size_t)B * L * C;
  __bf16* Vb  = Kb + (size_t)B * Lc * C;
  __bf16* AOb = Vb + (size_t)B * Lc * C;

  const dim3 blk(128);
  // projections (f32 in -> bf16 ws)
  gemm_wt_kernel<false, false><<<dim3(B * L / 64, C / 64), blk, 0, stream>>>(
      x, Wq, nullptr, Qb, B * L, C, C);
  gemm_wt_kernel<false, false><<<dim3(B * Lc / 64, C / 64), blk, 0, stream>>>(
      context, Wk, nullptr, Kb, B * Lc, C, C);
  gemm_wt_kernel<false, false><<<dim3(B * Lc / 64, C / 64), blk, 0, stream>>>(
      context, Wv, nullptr, Vb, B * Lc, C, C);
  // q: l2norm over head_dim * exp(min(scale_mul, log100))
  qscale_kernel<<<(B * L * N_HEADS) / 8, 256, 0, stream>>>(Qb, scale_mul, B * L, C);
  // k: l2norm over the Lc axis (faithful to source permute-before-normalize)
  kscale_kernel<<<dim3(C / 64, B), 256, 0, stream>>>(Kb, Lc, C);
  // flash attention with bias
  attn_kernel<<<dim3(L / 64, N_HEADS, B), blk, 0, stream>>>(
      Qb, Kb, Vb, attn_bias, AOb, B, L, Lc, C);
  // output projection + bias (bf16 in -> f32 out)
  gemm_wt_kernel<true, true><<<dim3(B * L / 64, C / 64), blk, 0, stream>>>(
      AOb, Wp, bp, out, B * L, C, C);
}
